// Mamba_21165598835546
// MI455X (gfx1250) — matrix operand order repaired and checked by reading, not compiled, not run on hardware
//
#include <hip/hip_runtime.h>
#include <hip/hip_bf16.h>
#include <stdint.h>

// ---------------------------------------------------------------------------
// Model constants (match reference)
// ---------------------------------------------------------------------------
#define D_MODEL  1024
#define D_INNER  2048
#define D_STATE  16
#define DT_RANK  64
#define D_CONV   4
#define N_LAYER  4
#define SEQ_B    2
#define SEQ_L    2048
#define ROWS     (SEQ_B * SEQ_L)          // 4096 token rows
#define XDBL_N   (DT_RANK + 2 * D_STATE)  // 96

typedef __attribute__((ext_vector_type(16))) __bf16 v16bf;
typedef __attribute__((ext_vector_type(8)))  __bf16 v8bf;
typedef __attribute__((ext_vector_type(8)))  float  v8f;

union Frag16 { v16bf v; v8bf h[2]; };

// Native RNE convert (v_cvt_*_bf16) instead of manual bit-twiddling.
__device__ __forceinline__ __bf16 f2bf(float f) { return (__bf16)f; }

// ---------------------------------------------------------------------------
// GEMM: C[M,N] = A[M,K](fp32,lda) * W[N,K](fp32, row-major)^T  (+ bias[N])
// fp32 -> bf16 conversion during LDS staging; compute = v_wmma_f32_16x16x32_bf16.
// Block tile 64x128x64, 256 threads = 8 waves, each wave owns a 32x32 output.
// Software pipelined: next K-slab's global loads issue before this slab's WMMAs.
// ---------------------------------------------------------------------------
constexpr int BM = 64, BN = 128, BK = 64, LDS_K = BK + 8; // stride 144B: conflict-free frags

__global__ __launch_bounds__(256)
void gemm_bf16_wmma(const float* __restrict__ A, const float* __restrict__ W,
                    const float* __restrict__ bias, float* __restrict__ C,
                    int M, int N, int K, int lda, int ldc)
{
  __shared__ __bf16 As[BM * LDS_K];   // 64 x 72 bf16
  __shared__ __bf16 Ws[BN * LDS_K];   // 128 x 72 bf16

  const int tid  = threadIdx.x;
  const int bm   = blockIdx.x * BM;
  const int bn   = blockIdx.y * BN;
  const int wv   = tid >> 5, lane = tid & 31;
  const int mw   = (wv & 1) * 32;      // wave's M offset inside block tile
  const int nw   = (wv >> 1) * 32;     // wave's N offset inside block tile
  const int lm   = lane & 15;          // row-within-16 per ISA fragment layout
  const int lsel = lane >> 4;          // half-wave selector

  v8f acc[2][2] = {};

  // staging map: thread covers row r = tid>>2, 4x float4 along K (64 cols)
  const int ra = tid >> 2;            // 0..63
  const int c4 = (tid & 3) * 4;       // 0,4,8,12 ; +q*16

  float4 aReg[4];       // A slab prefetch registers (row ra)
  float4 wReg[2][4];    // W slab prefetch registers (rows ra, ra+64)

  auto loadSlab = [&](int k0) {
    #pragma unroll
    for (int q = 0; q < 4; ++q)
      aReg[q] = *(const float4*)(A + (size_t)(bm + ra) * lda + k0 + c4 + q * 16);
    #pragma unroll
    for (int p = 0; p < 2; ++p) {
      int n = bn + ra + p * 64;
      #pragma unroll
      for (int q = 0; q < 4; ++q) {
        float4 v = make_float4(0.f, 0.f, 0.f, 0.f);
        if (n < N) v = *(const float4*)(W + (size_t)n * K + k0 + c4 + q * 16);
        wReg[p][q] = v;
      }
    }
  };

  auto storeSlab = [&]() {
    #pragma unroll
    for (int q = 0; q < 4; ++q) {
      __bf16* dst = &As[ra * LDS_K + c4 + q * 16];
      dst[0] = f2bf(aReg[q].x); dst[1] = f2bf(aReg[q].y);
      dst[2] = f2bf(aReg[q].z); dst[3] = f2bf(aReg[q].w);
    }
    #pragma unroll
    for (int p = 0; p < 2; ++p)
      #pragma unroll
      for (int q = 0; q < 4; ++q) {
        __bf16* dst = &Ws[(ra + p * 64) * LDS_K + c4 + q * 16];
        dst[0] = f2bf(wReg[p][q].x); dst[1] = f2bf(wReg[p][q].y);
        dst[2] = f2bf(wReg[p][q].z); dst[3] = f2bf(wReg[p][q].w);
      }
  };

  loadSlab(0);
  for (int k0 = 0; k0 < K; k0 += BK) {
    __syncthreads();              // previous slab's LDS reads complete
    storeSlab();
    __syncthreads();
    if (k0 + BK < K) loadSlab(k0 + BK);   // overlap next global loads with WMMAs

    // two 16x16x32 K-substeps per slab
    #pragma unroll
    for (int s = 0; s < 2; ++s) {
      const int kb = s * 32;
      Frag16 afr[2], bfr[2];
      #pragma unroll
      for (int i = 0; i < 2; ++i) {
        const __bf16* base = &As[(mw + i * 16 + lm) * LDS_K + kb];
        afr[i].h[0] = *(const v8bf*)(base + lsel * 8);        // K[0..7]  / K[8..15]
        afr[i].h[1] = *(const v8bf*)(base + lsel * 8 + 16);   // K[16..23]/ K[24..31]
      }
      #pragma unroll
      for (int j = 0; j < 2; ++j) {
        const __bf16* base = &Ws[(nw + j * 16 + lm) * LDS_K + kb];
        bfr[j].h[0] = *(const v8bf*)(base + lsel * 8);        // K[0..7]  / K[8..15]
        bfr[j].h[1] = *(const v8bf*)(base + lsel * 8 + 16);   // K[16..23]/ K[24..31]
      }
      #pragma unroll
      for (int i = 0; i < 2; ++i)
        #pragma unroll
        for (int j = 0; j < 2; ++j)
          acc[i][j] = __builtin_amdgcn_wmma_f32_16x16x32_bf16(
              false, afr[i].v, false, bfr[j].v, (short)0, acc[i][j], false, false);
    }
  }

  // ---- epilogue: C/D layout (col = lane&15, row = r + 8*(lane>=16)) ----
  #pragma unroll
  for (int i = 0; i < 2; ++i) {
    #pragma unroll
    for (int j = 0; j < 2; ++j) {
      int col = bn + nw + j * 16 + lm;
      if (col < N) {
        float badd = bias ? bias[col] : 0.f;
        #pragma unroll
        for (int r = 0; r < 8; ++r) {
          int row = bm + mw + i * 16 + r + lsel * 8;
          C[(size_t)row * ldc + col] = acc[i][j][r] + badd;
        }
      }
    }
  }
}

// ---------------------------------------------------------------------------
// Causal depthwise conv (K=4) over L + SiLU.  xz: [B,L,2*D_INNER], xp half.
// ---------------------------------------------------------------------------
__global__ __launch_bounds__(256)
void conv_silu_kernel(const float* __restrict__ xz, const float* __restrict__ cw,
                      const float* __restrict__ cb, float* __restrict__ xp)
{
  int idx = blockIdx.x * 256 + threadIdx.x;
  if (idx >= ROWS * D_INNER) return;
  int d  = idx % D_INNER;
  int bl = idx / D_INNER;
  int l  = bl % SEQ_L;
  float acc = cb[d];
  #pragma unroll
  for (int j = 0; j < D_CONV; ++j) {
    int ll = l - (D_CONV - 1) + j;
    if (ll >= 0)
      acc += cw[d * D_CONV + j] * xz[(size_t)(bl - (D_CONV - 1) + j) * (2 * D_INNER) + d];
  }
  xp[idx] = acc / (1.f + __expf(-acc));   // SiLU
}

// ---------------------------------------------------------------------------
// y *= silu(z), z = xz[..., D_INNER:]
// ---------------------------------------------------------------------------
__global__ __launch_bounds__(256)
void mul_silu_z_kernel(float* __restrict__ y, const float* __restrict__ xz)
{
  int idx = blockIdx.x * 256 + threadIdx.x;
  if (idx >= ROWS * D_INNER) return;
  int d  = idx % D_INNER;
  int bl = idx / D_INNER;
  float z = xz[(size_t)bl * (2 * D_INNER) + D_INNER + d];
  y[idx] *= z / (1.f + __expf(-z));
}

// ---------------------------------------------------------------------------
// Selective scan.  One thread = one channel d (state h[16] in VGPRs); each
// wave streams the per-step B/C vectors (32 floats) into a private LDS
// double-buffer with async global->LDS loads -> zero barriers in the L loop.
// ---------------------------------------------------------------------------
__device__ __forceinline__ uint32_t lds_off_of(const void* p) {
  return (uint32_t)(uintptr_t)p;       // low 32 bits of a generic LDS ptr = LDS offset
}
__device__ __forceinline__ void async_f32_to_lds(const float* g, uint32_t lds) {
  asm volatile("global_load_async_to_lds_b32 %0, %1, off"
               :: "v"(lds), "v"((uint64_t)(uintptr_t)g) : "memory");
}

__global__ __launch_bounds__(256)
void selective_scan_kernel(const float* __restrict__ dtp,   // [ROWS,D_INNER], bias pre-added
                           const float* __restrict__ xp,    // [ROWS,D_INNER]
                           const float* __restrict__ xdbl,  // [ROWS,96] (B at 64, C at 80)
                           const float* __restrict__ A_log, // [D_INNER,16]
                           const float* __restrict__ Dvec,  // [D_INNER]
                           float* __restrict__ ys)          // [ROWS,D_INNER]
{
  __shared__ float bc[8][2][32];       // per-wave private double buffer
  const int wv   = threadIdx.x >> 5;
  const int lane = threadIdx.x & 31;
  const int b    = blockIdx.x / (D_INNER / 256);
  const int d    = (blockIdx.x % (D_INNER / 256)) * 256 + threadIdx.x;

  float Arow[D_STATE];
  #pragma unroll
  for (int n = 0; n < D_STATE; ++n) Arow[n] = -__expf(A_log[(size_t)d * D_STATE + n]);
  const float Dv = Dvec[d];
  float hst[D_STATE];
  #pragma unroll
  for (int n = 0; n < D_STATE; ++n) hst[n] = 0.f;

  // prefetch step 0's B/C (lane i fetches xdbl[.., 64+i])
  async_f32_to_lds(xdbl + (size_t)(b * SEQ_L) * XDBL_N + DT_RANK + lane,
                   lds_off_of(&bc[wv][0][lane]));

  for (int l = 0; l < SEQ_L; ++l) {
    const int buf = l & 1;
    if (l + 1 < SEQ_L) {
      async_f32_to_lds(xdbl + (size_t)(b * SEQ_L + l + 1) * XDBL_N + DT_RANK + lane,
                       lds_off_of(&bc[wv][buf ^ 1][lane]));
      asm volatile("s_wait_asynccnt 0x1" ::: "memory");  // step l's buffer complete
    } else {
      asm volatile("s_wait_asynccnt 0x0" ::: "memory");
    }
    const size_t idx = (size_t)(b * SEQ_L + l) * D_INNER + d;
    float draw  = dtp[idx];
    float delta = (draw > 20.f) ? draw : log1pf(__expf(draw));  // softplus
    float u  = xp[idx];
    float du = delta * u;
    const float* Bt = &bc[wv][buf][0];
    const float* Ct = &bc[wv][buf][16];
    float y = 0.f;
    #pragma unroll
    for (int n = 0; n < D_STATE; ++n) {
      float dA = __expf(delta * Arow[n]);
      hst[n] = dA * hst[n] + du * Bt[n];
      y += hst[n] * Ct[n];
    }
    ys[idx] = y + u * Dv;
  }
}

// ---------------------------------------------------------------------------
// RMSNorm helpers (row = 1024 wide, one block per row)
// ---------------------------------------------------------------------------
__global__ __launch_bounds__(256)
void rmsnorm_add_kernel(const float* __restrict__ y, const float* __restrict__ w,
                        float* __restrict__ h)
{
  __shared__ float red[256];
  const int row = blockIdx.x, tid = threadIdx.x;
  const float* yr = y + (size_t)row * D_MODEL;
  float s = 0.f;
  for (int c = tid; c < D_MODEL; c += 256) { float v = yr[c]; s += v * v; }
  red[tid] = s; __syncthreads();
  for (int st = 128; st > 0; st >>= 1) {
    if (tid < st) red[tid] += red[tid + st];
    __syncthreads();
  }
  float r = rsqrtf(red[0] / (float)D_MODEL + 1e-5f);
  for (int c = tid; c < D_MODEL; c += 256)
    h[(size_t)row * D_MODEL + c] += yr[c] * r * w[c];
}

__global__ __launch_bounds__(256)
void rmsnorm_kernel(const float* __restrict__ x, const float* __restrict__ w,
                    float* __restrict__ o)
{
  __shared__ float red[256];
  const int row = blockIdx.x, tid = threadIdx.x;
  const float* xr = x + (size_t)row * D_MODEL;
  float s = 0.f;
  for (int c = tid; c < D_MODEL; c += 256) { float v = xr[c]; s += v * v; }
  red[tid] = s; __syncthreads();
  for (int st = 128; st > 0; st >>= 1) {
    if (tid < st) red[tid] += red[tid + st];
    __syncthreads();
  }
  float r = rsqrtf(red[0] / (float)D_MODEL + 1e-5f);
  for (int c = tid; c < D_MODEL; c += 256)
    o[(size_t)row * D_MODEL + c] = xr[c] * r * w[c];
}

// ---------------------------------------------------------------------------
// Orchestration
// ---------------------------------------------------------------------------
extern "C" void kernel_launch(void* const* d_in, const int* in_sizes, int n_in,
                              void* d_out, int out_size, void* d_ws, size_t ws_size,
                              hipStream_t stream)
{
  (void)in_sizes; (void)n_in; (void)out_size; (void)ws_size;
  const float* x       = (const float*)d_in[0];
  const float* emb_w   = (const float*)d_in[1];
  const float* emb_b   = (const float*)d_in[2];
  const float* in_w    = (const float*)d_in[3];
  const float* conv_w  = (const float*)d_in[4];
  const float* conv_b  = (const float*)d_in[5];
  const float* xproj_w = (const float*)d_in[6];
  const float* dt_w    = (const float*)d_in[7];
  const float* dt_b    = (const float*)d_in[8];
  const float* A_log   = (const float*)d_in[9];
  const float* Dp      = (const float*)d_in[10];
  const float* out_w   = (const float*)d_in[11];
  const float* norm_w  = (const float*)d_in[12];
  const float* head_w  = (const float*)d_in[13];
  float* out = (float*)d_out;

  char* ws = (char*)d_ws;
  size_t off = 0;
  auto alloc = [&](size_t elems) { float* p = (float*)(ws + off); off += elems * sizeof(float); return p; };
  float* h    = alloc((size_t)ROWS * D_MODEL);        // residual stream
  float* xz   = alloc((size_t)ROWS * 2 * D_INNER);
  float* xp   = alloc((size_t)ROWS * D_INNER);
  float* xdbl = alloc((size_t)ROWS * XDBL_N);
  float* dtp  = alloc((size_t)ROWS * D_INNER);
  float* ysc  = alloc((size_t)ROWS * D_INNER);
  float* tmp  = alloc((size_t)ROWS * D_MODEL);        // per-layer out_proj / final norm

  auto gemm = [&](const float* A, const float* W, const float* bias, float* C,
                  int M, int N, int K, int lda, int ldc) {
    dim3 grid(M / BM, (N + BN - 1) / BN);
    gemm_bf16_wmma<<<grid, 256, 0, stream>>>(A, W, bias, C, M, N, K, lda, ldc);
  };

  // h = x @ emb_w^T + emb_b
  gemm(x, emb_w, emb_b, h, ROWS, D_MODEL, D_MODEL, D_MODEL, D_MODEL);

  const int ew = ROWS * D_INNER;
  for (int i = 0; i < N_LAYER; ++i) {
    const float* inw = in_w    + (size_t)i * 2 * D_INNER * D_MODEL;
    const float* cw  = conv_w  + (size_t)i * D_INNER * D_CONV;
    const float* cb  = conv_b  + (size_t)i * D_INNER;
    const float* xpw = xproj_w + (size_t)i * XDBL_N * D_INNER;
    const float* dtw = dt_w    + (size_t)i * D_INNER * DT_RANK;
    const float* dtb = dt_b    + (size_t)i * D_INNER;
    const float* Al  = A_log   + (size_t)i * D_INNER * D_STATE;
    const float* Dv  = Dp      + (size_t)i * D_INNER;
    const float* ow  = out_w   + (size_t)i * D_MODEL * D_INNER;

    gemm(h, inw, nullptr, xz, ROWS, 2 * D_INNER, D_MODEL, D_MODEL, 2 * D_INNER);
    conv_silu_kernel<<<(ew + 255) / 256, 256, 0, stream>>>(xz, cw, cb, xp);
    gemm(xp, xpw, nullptr, xdbl, ROWS, XDBL_N, D_INNER, D_INNER, XDBL_N);
    gemm(xdbl, dtw, dtb, dtp, ROWS, D_INNER, DT_RANK, XDBL_N, D_INNER);
    selective_scan_kernel<<<SEQ_B * (D_INNER / 256), 256, 0, stream>>>(dtp, xp, xdbl, Al, Dv, ysc);
    mul_silu_z_kernel<<<(ew + 255) / 256, 256, 0, stream>>>(ysc, xz);
    gemm(ysc, ow, nullptr, tmp, ROWS, D_MODEL, D_INNER, D_INNER, D_MODEL);
    rmsnorm_add_kernel<<<ROWS, 256, 0, stream>>>(tmp, norm_w, h);
  }
  rmsnorm_kernel<<<ROWS, 256, 0, stream>>>(h, norm_w, tmp);
  gemm(tmp, head_w, nullptr, out, ROWS, D_MODEL, D_MODEL, D_MODEL, D_MODEL);
}
